// ImageStreamBlock_30623116821345
// MI455X (gfx1250) — compile-verified, matrix operand order checked by reading, not run on hardware
//
#include <hip/hip_runtime.h>
#include <math.h>

typedef __bf16  bf16_t;
typedef __bf16  v16bf __attribute__((ext_vector_type(16)));
typedef float   v8f   __attribute__((ext_vector_type(8)));

#define B0   1
#define LI   2048
#define LT   256
#define LL   2304          // LI + LT
#define HH   1024
#define NHH  16
#define HDD  64
#define MLPD 4096

// ---------------------------------------------------------------------------
// Fragment loaders per CDNA5 ISA 7.12.2 (16-bit A 16x32, B symmetric, C/D f32)
// ---------------------------------------------------------------------------
__device__ __forceinline__ v16bf load_A_frag(const bf16_t* A, int lda,
                                             int row0, int k0, int lane) {
    // A row-major [M x lda]; tile rows row0..row0+15, K cols k0..k0+31
    int m = lane & 15, h = lane >> 4;
    v16bf a;
#pragma unroll
    for (int j = 0; j < 8; ++j) {
        int kb = (j < 4) ? (h * 8 + 2 * j) : (16 + h * 8 + 2 * (j - 4));
        const bf16_t* p = A + (size_t)(row0 + m) * lda + (k0 + kb);
        a[2 * j]     = p[0];
        a[2 * j + 1] = p[1];
    }
    return a;
}

__device__ __forceinline__ v16bf load_Bt_frag(const bf16_t* Bt, int ldt,
                                              int k0, int col0, int lane) {
    // Logical B is [K x N]; stored transposed Bt[N][ldt] so pairs along K are
    // contiguous.  B[k][n] = Bt[(col0+n)*ldt + k]
    int n = lane & 15, h = lane >> 4;
    v16bf b;
#pragma unroll
    for (int j = 0; j < 8; ++j) {
        int kb = (j < 4) ? (h * 8 + 2 * j) : (16 + h * 8 + 2 * (j - 4));
        const bf16_t* p = Bt + (size_t)(col0 + n) * ldt + (k0 + kb);
        b[2 * j]     = p[0];
        b[2 * j + 1] = p[1];
    }
    return b;
}

__device__ __forceinline__ float gelu_tanh(float x) {
    float x3 = x * x * x;
    return 0.5f * x * (1.0f + tanhf(0.7978845608028654f * (x + 0.044715f * x3)));
}

// ---------------------------------------------------------------------------
// Small helper kernels
// ---------------------------------------------------------------------------
__global__ void k_silu(const float* v, float* sv) {
    int i = blockIdx.x * 256 + threadIdx.x;
    if (i < HH) { float x = v[i]; sv[i] = x / (1.0f + __expf(-x)); }
}

__global__ void k_mod(const float* sv, const float* W, const float* b, float* out) {
    int j = blockIdx.x * 256 + threadIdx.x;          // j < 6*HH
    float acc = b[j];
    for (int d = 0; d < HH; ++d) acc += sv[d] * W[(size_t)d * (6 * HH) + j];
    out[j] = acc;
}

__global__ void k_cvt_t(const float* W, bf16_t* Wt, int K, int N) {
    size_t idx = (size_t)blockIdx.x * 256 + threadIdx.x;
    if (idx >= (size_t)K * N) return;
    int k = (int)(idx / N), n = (int)(idx % N);
    Wt[(size_t)n * K + k] = (bf16_t)W[idx];
}

// LayerNorm (no affine) + modulation: y = (1+scale)*ln(x) + shift   -> bf16
__global__ void k_ln_mod(const float* x, bf16_t* y,
                         const float* shift, const float* scale) {
    int row = blockIdx.x, tid = threadIdx.x;
    const float* xr = x + (size_t)row * HH;
    __shared__ float red[256];
    float v[4]; float s = 0.f;
#pragma unroll
    for (int i = 0; i < 4; ++i) { v[i] = xr[tid + 256 * i]; s += v[i]; }
    red[tid] = s; __syncthreads();
    for (int off = 128; off > 0; off >>= 1) {
        if (tid < off) red[tid] += red[tid + off];
        __syncthreads();
    }
    float mean = red[0] * (1.0f / HH);
    __syncthreads();
    float q = 0.f;
#pragma unroll
    for (int i = 0; i < 4; ++i) { float d = v[i] - mean; q += d * d; }
    red[tid] = q; __syncthreads();
    for (int off = 128; off > 0; off >>= 1) {
        if (tid < off) red[tid] += red[tid + off];
        __syncthreads();
    }
    float inv = rsqrtf(red[0] * (1.0f / HH) + 1e-6f);
#pragma unroll
    for (int i = 0; i < 4; ++i) {
        int c = tid + 256 * i;
        float nv = (v[i] - mean) * inv;
        y[(size_t)row * HH + c] = (bf16_t)((1.0f + scale[c]) * nv + shift[c]);
    }
}

// ---------------------------------------------------------------------------
// Generic bf16 WMMA GEMM: out[M,N] = act(A[M,K] * B[K,N] + bias), epilogue
// optionally  resid + gate*out.  One wave computes a 16x64 strip of C.
// Next K-step's cache lines are prefetched (global_prefetch_b8) so WGP$ fill
// overlaps the current WMMAs.
// ---------------------------------------------------------------------------
__global__ void k_gemm(const bf16_t* A, const bf16_t* Bt, const float* bias,
                       float* outF, bf16_t* outB,
                       const float* resid, const float* gate,
                       int M, int N, int K, int act) {
    int lane = threadIdx.x;
    int row0 = blockIdx.y * 16;
    int col0 = blockIdx.x * 64;
    int m = lane & 15;
    v8f acc[4] = {};
    for (int k0 = 0; k0 < K; k0 += 32) {
        if (k0 + 32 < K) {
            // prefetch next K tile: one lane-distinct byte per 64B line
            __builtin_prefetch(A + (size_t)(row0 + m) * K + (k0 + 32) + ((lane >> 4) << 4), 0, 3);
#pragma unroll
            for (int t = 0; t < 4; ++t)
                __builtin_prefetch(Bt + (size_t)(col0 + 16 * t + m) * K + (k0 + 32) + ((lane >> 4) << 4), 0, 3);
        }
        v16bf a = load_A_frag(A, K, row0, k0, lane);
#pragma unroll
        for (int t = 0; t < 4; ++t) {
            v16bf b = load_Bt_frag(Bt, K, k0, col0 + 16 * t, lane);
            acc[t] = __builtin_amdgcn_wmma_f32_16x16x32_bf16(
                false, a, false, b, (short)0, acc[t], false, false);
        }
    }
    int nl = lane & 15;
#pragma unroll
    for (int t = 0; t < 4; ++t) {
#pragma unroll
        for (int r = 0; r < 8; ++r) {
            int mm = row0 + ((lane < 16) ? r : r + 8);
            int n  = col0 + 16 * t + nl;
            float v = acc[t][r] + (bias ? bias[n] : 0.0f);
            if (act == 1) v = gelu_tanh(v);
            if (gate)     v = resid[(size_t)mm * N + n] + gate[n] * v;
            if (outF) outF[(size_t)mm * N + n] = v;
            if (outB) outB[(size_t)mm * N + n] = (bf16_t)v;
        }
    }
}

// ---------------------------------------------------------------------------
// RMS-norm + RoPE + repack: qkv f32 -> Q,K [NH][L][HD] bf16 (q pre-scaled by
// 1/sqrt(HD)), V transposed [NH][HD][L] bf16 (so P*V B-pairs are contiguous).
// ---------------------------------------------------------------------------
__global__ void k_qk_rope(const float* qkv_txt, const float* qkv_img,
                          const float* tq_s, const float* tk_s,
                          const float* iq_s, const float* ik_s,
                          const float* pe,
                          bf16_t* Qb, bf16_t* Kb, bf16_t* Vt) {
    int l = blockIdx.x, h = blockIdx.y, lane = threadIdx.x;
    const float *src, *qs, *ks;
    if (l < LT) { src = qkv_txt + (size_t)l * (3 * HH);       qs = tq_s; ks = tk_s; }
    else        { src = qkv_img + (size_t)(l - LT) * (3 * HH); qs = iq_s; ks = ik_s; }
    int d0 = 2 * lane, d1 = 2 * lane + 1;
    float q0 = src[0 * HH + h * HDD + d0], q1 = src[0 * HH + h * HDD + d1];
    float k0 = src[1 * HH + h * HDD + d0], k1 = src[1 * HH + h * HDD + d1];
    float v0 = src[2 * HH + h * HDD + d0], v1 = src[2 * HH + h * HDD + d1];

    float sq = q0 * q0 + q1 * q1;
    float sk = k0 * k0 + k1 * k1;
#pragma unroll
    for (int off = 1; off < 32; off <<= 1) {
        sq += __shfl_xor(sq, off, 32);
        sk += __shfl_xor(sk, off, 32);
    }
    float qr = rsqrtf(sq * (1.0f / HDD) + 1e-6f);
    float kr = rsqrtf(sk * (1.0f / HDD) + 1e-6f);
    q0 *= qr * qs[d0]; q1 *= qr * qs[d1];
    k0 *= kr * ks[d0]; k1 *= kr * ks[d1];

    const float* per = pe + (size_t)l * 128 + lane * 4;  // (L, 32, 2, 2)
    float c = per[0], s = per[2];
    float Q0 = (c * q0 - s * q1) * 0.125f;   // fold 1/sqrt(64)
    float Q1 = (s * q0 + c * q1) * 0.125f;
    float K0 = c * k0 - s * k1;
    float K1 = s * k0 + c * k1;

    size_t base = ((size_t)h * LL + l) * HDD;
    Qb[base + d0] = (bf16_t)Q0; Qb[base + d1] = (bf16_t)Q1;
    Kb[base + d0] = (bf16_t)K0; Kb[base + d1] = (bf16_t)K1;
    Vt[((size_t)h * HDD + d0) * LL + l] = (bf16_t)v0;
    Vt[((size_t)h * HDD + d1) * LL + l] = (bf16_t)v1;
}

// ---------------------------------------------------------------------------
// Flash attention: one wave per (head, 16-query tile); online softmax over
// 32-key blocks; Q*K^T and P*V both via v_wmma_f32_16x16x32_bf16.
// ---------------------------------------------------------------------------
__global__ void k_attn(const bf16_t* Q, const bf16_t* Kc, const bf16_t* Vt,
                       bf16_t* out /* [L][NH*HD] */) {
    int head = blockIdx.y;
    int q0   = blockIdx.x * 16;
    int lane = threadIdx.x;
    const bf16_t* Qh = Q  + (size_t)head * LL * HDD;
    const bf16_t* Kh = Kc + (size_t)head * LL * HDD;
    const bf16_t* Vh = Vt + (size_t)head * HDD * LL;

    v16bf qa0 = load_A_frag(Qh, HDD, q0, 0,  lane);
    v16bf qa1 = load_A_frag(Qh, HDD, q0, 32, lane);

    v8f o[4] = {};
    float mrow[8], lrow[8];
#pragma unroll
    for (int r = 0; r < 8; ++r) { mrow[r] = -3.0e38f; lrow[r] = 0.f; }

    __shared__ bf16_t pbuf[16 * 32];
    int nl = lane & 15;

    for (int kb = 0; kb < LL; kb += 32) {
        if (kb + 32 < LL) {
            // prefetch next 32 keys (row-major [L][64] bf16 = two lines/row)
            __builtin_prefetch(Kh + (size_t)(kb + 32 + lane) * HDD, 0, 3);
            // prefetch next V columns ([HD][L] bf16: 2 dim rows per lane)
            __builtin_prefetch(Vh + (size_t)(lane) * LL + kb + 32, 0, 3);
            __builtin_prefetch(Vh + (size_t)(lane + 32) * LL + kb + 32, 0, 3);
        }
        v8f s[2] = {};
#pragma unroll
        for (int t = 0; t < 2; ++t) {
            v16bf b0 = load_Bt_frag(Kh, HDD, 0,  kb + 16 * t, lane);
            s[t] = __builtin_amdgcn_wmma_f32_16x16x32_bf16(
                false, qa0, false, b0, (short)0, s[t], false, false);
            v16bf b1 = load_Bt_frag(Kh, HDD, 32, kb + 16 * t, lane);
            s[t] = __builtin_amdgcn_wmma_f32_16x16x32_bf16(
                false, qa1, false, b1, (short)0, s[t], false, false);
        }
        // online softmax per row (rows live in r / lane-half, cols in lanes)
#pragma unroll
        for (int r = 0; r < 8; ++r) {
            float mx = fmaxf(s[0][r], s[1][r]);
#pragma unroll
            for (int off = 1; off < 16; off <<= 1) mx = fmaxf(mx, __shfl_xor(mx, off, 32));
            float mnew  = fmaxf(mrow[r], mx);
            float alpha = __expf(mrow[r] - mnew);
            mrow[r] = mnew;
            float p0 = __expf(s[0][r] - mnew);
            float p1 = __expf(s[1][r] - mnew);
            s[0][r] = p0; s[1][r] = p1;
            float rs = p0 + p1;
#pragma unroll
            for (int off = 1; off < 16; off <<= 1) rs += __shfl_xor(rs, off, 32);
            lrow[r] = lrow[r] * alpha + rs;
#pragma unroll
            for (int t = 0; t < 4; ++t) o[t][r] *= alpha;
        }
        // bounce P through LDS to convert C-layout -> A-fragment layout
#pragma unroll
        for (int r = 0; r < 8; ++r) {
            int mm = (lane < 16) ? r : r + 8;
            pbuf[mm * 32 + nl]      = (bf16_t)s[0][r];
            pbuf[mm * 32 + 16 + nl] = (bf16_t)s[1][r];
        }
        __syncthreads();
        v16bf pa = load_A_frag(pbuf, 32, 0, 0, lane);
        __syncthreads();
#pragma unroll
        for (int t = 0; t < 4; ++t) {
            v16bf vb = load_Bt_frag(Vh, LL, kb, 16 * t, lane);
            o[t] = __builtin_amdgcn_wmma_f32_16x16x32_bf16(
                false, pa, false, vb, (short)0, o[t], false, false);
        }
    }
    // normalize + write [token][head*64+dim] bf16
#pragma unroll
    for (int t = 0; t < 4; ++t) {
#pragma unroll
        for (int r = 0; r < 8; ++r) {
            int mm = (lane < 16) ? r : r + 8;
            int token = q0 + mm;
            int dim   = 16 * t + nl;
            out[(size_t)token * (NHH * HDD) + head * HDD + dim] =
                (bf16_t)(o[t][r] / lrow[r]);
        }
    }
}

// ---------------------------------------------------------------------------
// Orchestration
// ---------------------------------------------------------------------------
extern "C" void kernel_launch(void* const* d_in, const int* in_sizes, int n_in,
                              void* d_out, int out_size, void* d_ws, size_t ws_size,
                              hipStream_t stream) {
    const float* img        = (const float*)d_in[0];
    const float* txt        = (const float*)d_in[1];
    const float* pe         = (const float*)d_in[2];
    const float* vec        = (const float*)d_in[3];
    const float* img_mod_w  = (const float*)d_in[4];
    const float* img_mod_b  = (const float*)d_in[5];
    const float* txt_mod_w  = (const float*)d_in[6];
    const float* txt_mod_b  = (const float*)d_in[7];
    const float* img_qkv_w  = (const float*)d_in[8];
    const float* img_qkv_b  = (const float*)d_in[9];
    const float* img_q_s    = (const float*)d_in[10];
    const float* img_k_s    = (const float*)d_in[11];
    const float* img_proj_w = (const float*)d_in[12];
    const float* img_proj_b = (const float*)d_in[13];
    const float* txt_qkv_w  = (const float*)d_in[14];
    const float* txt_qkv_b  = (const float*)d_in[15];
    const float* txt_q_s    = (const float*)d_in[16];
    const float* txt_k_s    = (const float*)d_in[17];
    const float* mlp_w1     = (const float*)d_in[18];
    const float* mlp_b1     = (const float*)d_in[19];
    const float* mlp_w2     = (const float*)d_in[20];
    const float* mlp_b2     = (const float*)d_in[21];

    size_t off = 0;
    auto alloc = [&](size_t bytes) {
        void* p = (char*)d_ws + off;
        off += (bytes + 255) & ~(size_t)255;
        return p;
    };
    float*  sv       = (float*)alloc(HH * 4);
    float*  imod     = (float*)alloc(6 * HH * 4);
    float*  tmod     = (float*)alloc(6 * HH * 4);
    bf16_t* w_qkvi_t = (bf16_t*)alloc((size_t)3 * HH * HH * 2);
    bf16_t* w_qkvt_t = (bf16_t*)alloc((size_t)3 * HH * HH * 2);
    bf16_t* w_proj_t = (bf16_t*)alloc((size_t)HH * HH * 2);
    bf16_t* w_mlp1_t = (bf16_t*)alloc((size_t)HH * MLPD * 2);
    bf16_t* w_mlp2_t = (bf16_t*)alloc((size_t)MLPD * HH * 2);
    bf16_t* img_ln   = (bf16_t*)alloc((size_t)LI * HH * 2);
    bf16_t* txt_ln   = (bf16_t*)alloc((size_t)LT * HH * 2);
    float*  qkv_img  = (float*)alloc((size_t)LI * 3 * HH * 4);
    float*  qkv_txt  = (float*)alloc((size_t)LT * 3 * HH * 4);
    bf16_t* Qb       = (bf16_t*)alloc((size_t)NHH * LL * HDD * 2);
    bf16_t* Kb       = (bf16_t*)alloc((size_t)NHH * LL * HDD * 2);
    bf16_t* Vt       = (bf16_t*)alloc((size_t)NHH * HDD * LL * 2);
    bf16_t* attn     = (bf16_t*)alloc((size_t)LL * HH * 2);
    float*  img2     = (float*)alloc((size_t)LI * HH * 4);
    bf16_t* hln      = (bf16_t*)alloc((size_t)LI * HH * 2);
    bf16_t* hact     = (bf16_t*)alloc((size_t)LI * MLPD * 2);

    // 1) modulation vectors
    k_silu<<<4, 256, 0, stream>>>(vec, sv);
    k_mod<<<24, 256, 0, stream>>>(sv, img_mod_w, img_mod_b, imod);
    k_mod<<<24, 256, 0, stream>>>(sv, txt_mod_w, txt_mod_b, tmod);

    // 2) weight fp32 -> bf16 transposed for contiguous B-fragment pairs
    k_cvt_t<<<(3 * HH * HH + 255) / 256, 256, 0, stream>>>(img_qkv_w,  w_qkvi_t, HH, 3 * HH);
    k_cvt_t<<<(3 * HH * HH + 255) / 256, 256, 0, stream>>>(txt_qkv_w,  w_qkvt_t, HH, 3 * HH);
    k_cvt_t<<<(HH * HH     + 255) / 256, 256, 0, stream>>>(img_proj_w, w_proj_t, HH, HH);
    k_cvt_t<<<(HH * MLPD   + 255) / 256, 256, 0, stream>>>(mlp_w1,     w_mlp1_t, HH, MLPD);
    k_cvt_t<<<(MLPD * HH   + 255) / 256, 256, 0, stream>>>(mlp_w2,     w_mlp2_t, MLPD, HH);

    // 3) layernorm + modulation (shift1/scale1 = chunks 0,1)
    k_ln_mod<<<LI, 256, 0, stream>>>(img, img_ln, imod, imod + HH);
    k_ln_mod<<<LT, 256, 0, stream>>>(txt, txt_ln, tmod, tmod + HH);

    // 4) QKV projections (WMMA)
    k_gemm<<<dim3(3 * HH / 64, LI / 16), 32, 0, stream>>>(
        img_ln, w_qkvi_t, img_qkv_b, qkv_img, nullptr, nullptr, nullptr,
        LI, 3 * HH, HH, 0);
    k_gemm<<<dim3(3 * HH / 64, LT / 16), 32, 0, stream>>>(
        txt_ln, w_qkvt_t, txt_qkv_b, qkv_txt, nullptr, nullptr, nullptr,
        LT, 3 * HH, HH, 0);

    // 5) RMS norm + RoPE + repack (txt tokens first)
    k_qk_rope<<<dim3(LL, NHH), 32, 0, stream>>>(
        qkv_txt, qkv_img, txt_q_s, txt_k_s, img_q_s, img_k_s, pe, Qb, Kb, Vt);

    // 6) flash attention (WMMA for QK^T and PV)
    k_attn<<<dim3(LL / 16, NHH), 32, 0, stream>>>(Qb, Kb, Vt, attn);

    // 7) output projection on img tokens, fused residual + gate1 (chunk 2)
    k_gemm<<<dim3(HH / 64, LI / 16), 32, 0, stream>>>(
        attn + (size_t)LT * HH, w_proj_t, img_proj_b, img2, nullptr,
        img, imod + 2 * HH, LI, HH, HH, 0);

    // 8) LN + modulation for MLP (shift2/scale2 = chunks 3,4)
    k_ln_mod<<<LI, 256, 0, stream>>>(img2, hln, imod + 3 * HH, imod + 4 * HH);

    // 9) MLP up (GELU fused, bf16 out feeds next GEMM)
    k_gemm<<<dim3(MLPD / 64, LI / 16), 32, 0, stream>>>(
        hln, w_mlp1_t, mlp_b1, nullptr, hact, nullptr, nullptr,
        LI, MLPD, HH, 1);

    // 10) MLP down, fused residual + gate2 (chunk 5) -> final f32 output
    k_gemm<<<dim3(HH / 64, LI / 16), 32, 0, stream>>>(
        hact, w_mlp2_t, mlp_b2, (float*)d_out, nullptr,
        img2, imod + 5 * HH, LI, HH, MLPD, 0);
}
